// FANBlock_29111288333035
// MI455X (gfx1250) — matrix operand order repaired and checked by reading, not compiled, not run on hardware
//
#include <hip/hip_runtime.h>
#include <math.h>

typedef _Float16 v16h __attribute__((ext_vector_type(16)));
typedef _Float16 v8h  __attribute__((ext_vector_type(8)));
typedef float    v8f  __attribute__((ext_vector_type(8)));

constexpr int CB    = 8;     // batch
constexpr int CN    = 1024;  // tokens
constexpr int CD    = 768;   // dim
constexpr int CH    = 8;     // heads
constexpr int CHD   = 96;    // head dim
constexpr int CHID  = 3072;  // mlp hidden
constexpr int CROWS = CB * CN;

// ---------------------------------------------------------------------------
// CDNA5 async global->LDS DMA (ASYNCcnt-tracked), per ISA 15.18.3 op 98.
// ---------------------------------------------------------------------------
typedef __attribute__((address_space(3))) const void* lds_cptr_t;

__device__ __forceinline__ void async_copy_b128(const void* gsrc, const void* ldst) {
  unsigned loff = (unsigned)(size_t)(lds_cptr_t)ldst;  // LDS byte offset
  asm volatile("global_load_async_to_lds_b128 %0, %1, off"
               :: "v"(loff), "v"(gsrc) : "memory");
}
__device__ __forceinline__ void wait_async0() {
  asm volatile("s_wait_asynccnt 0x0" ::: "memory");
}

// ---------------------------------------------------------------------------
// f32 -> f16 conversion (weights)
// ---------------------------------------------------------------------------
__global__ __launch_bounds__(256) void cvt_f32_f16(const float* __restrict__ s,
                                                   _Float16* __restrict__ d, int n) {
  int i = blockIdx.x * 256 + threadIdx.x;
  if (i < n) d[i] = (_Float16)s[i];
}

// ---------------------------------------------------------------------------
// Row LayerNorm (fp32 in, f16 out)
// ---------------------------------------------------------------------------
__global__ __launch_bounds__(256) void ln_rows(const float* __restrict__ xin,
                                               const float* __restrict__ w,
                                               const float* __restrict__ bb,
                                               _Float16* __restrict__ outH) {
  __shared__ float red[256];
  const int row = blockIdx.x, tid = threadIdx.x;
  const float* xr = xin + (size_t)row * CD;
  float s = 0.f, s2 = 0.f;
  for (int c = tid; c < CD; c += 256) { float v = xr[c]; s += v; s2 += v * v; }
  red[tid] = s; __syncthreads();
  for (int st = 128; st > 0; st >>= 1) { if (tid < st) red[tid] += red[tid + st]; __syncthreads(); }
  float mean = red[0] / CD; __syncthreads();
  red[tid] = s2; __syncthreads();
  for (int st = 128; st > 0; st >>= 1) { if (tid < st) red[tid] += red[tid + st]; __syncthreads(); }
  float var = red[0] / CD - mean * mean;
  float rs = rsqrtf(var + 1e-5f);
  for (int c = tid; c < CD; c += 256)
    outH[(size_t)row * CD + c] = (_Float16)((xr[c] - mean) * rs * w[c] + bb[c]);
}

// ---------------------------------------------------------------------------
// WMMA GEMM: out[M,N] = A[M,K] * W[N,K]^T  (+bias) (+resid + gamma*acc)
// block = 256 threads (8 waves), tile 128x128, K-step 32, async-LDS staged.
// ---------------------------------------------------------------------------
__global__ __launch_bounds__(256) void gemm_wmma(
    const _Float16* __restrict__ A, const _Float16* __restrict__ W,
    int M, int N, int K,
    const float* __restrict__ bias,
    const float* __restrict__ resid, const float* __restrict__ gamma,
    _Float16* __restrict__ outH, float* __restrict__ outF) {
  (void)M;
  __shared__ _Float16 Abuf[128 * 32];
  __shared__ _Float16 Bbuf[128 * 32];
  const int nb = N >> 7;
  const int m0 = (blockIdx.x / nb) << 7;
  const int n0 = (blockIdx.x % nb) << 7;
  const int tid = threadIdx.x;
  const int w = tid >> 5, l = tid & 31;
  const int wm = w & 3, wn = w >> 2;       // 4 x 2 wave grid: 32-row x 64-col per wave
  const int l15 = l & 15, lh = (l >> 4) & 1;

  v8f acc[2][4];
#pragma unroll
  for (int mt = 0; mt < 2; ++mt)
#pragma unroll
    for (int nt = 0; nt < 4; ++nt)
#pragma unroll
      for (int r = 0; r < 8; ++r) acc[mt][nt][r] = 0.f;

  for (int kt = 0; kt < K; kt += 32) {
    // async DMA the 128x32 A tile and 128x32 B tile straight into LDS
#pragma unroll
    for (int it = 0; it < 2; ++it) {
      int c = tid + (it << 8);           // 0..511 chunk id
      int r = c >> 2, s = c & 3;         // 128 rows x 4 chunks of 8 halves
      async_copy_b128(A + (size_t)(m0 + r) * K + kt + s * 8, Abuf + r * 32 + s * 8);
      async_copy_b128(W + (size_t)(n0 + r) * K + kt + s * 8, Bbuf + r * 32 + s * 8);
    }
    if (kt + 32 < K) {  // global_prefetch_b8 of next K-tile, WGP scope
      __builtin_prefetch(A + (size_t)(m0 + (tid >> 1)) * K + kt + 32, 0, 3);
      __builtin_prefetch(W + (size_t)(n0 + (tid >> 1)) * K + kt + 32, 0, 3);
    }
    wait_async0();
    __syncthreads();

    v16h af[2], bf[4];
#pragma unroll
    for (int mt = 0; mt < 2; ++mt) {
      const _Float16* p = Abuf + (wm * 32 + mt * 16 + l15) * 32 + lh * 16;
      v8h lo = *(const v8h*)p;
      v8h hi = *(const v8h*)(p + 8);
      v16h a;
#pragma unroll
      for (int i = 0; i < 8; ++i) { a[i] = lo[i]; a[i + 8] = hi[i]; }
      af[mt] = a;
    }
#pragma unroll
    for (int nt = 0; nt < 4; ++nt)
      bf[nt] = *(const v16h*)(Bbuf + (wn * 64 + nt * 16 + l15) * 32 + lh * 16);
#pragma unroll
    for (int mt = 0; mt < 2; ++mt)
#pragma unroll
      for (int nt = 0; nt < 4; ++nt)
        acc[mt][nt] = __builtin_amdgcn_wmma_f32_16x16x32_f16(
            false, af[mt], false, bf[nt], (short)0, acc[mt][nt], false, false);
    __syncthreads();
  }

#pragma unroll
  for (int mt = 0; mt < 2; ++mt)
#pragma unroll
    for (int nt = 0; nt < 4; ++nt) {
      const int col = n0 + wn * 64 + nt * 16 + l15;
      const int rowb = m0 + wm * 32 + mt * 16 + lh * 8;
#pragma unroll
      for (int r = 0; r < 8; ++r) {
        float v = acc[mt][nt][r];
        const size_t o = (size_t)(rowb + r) * N + col;
        if (bias) v += bias[col];
        if (resid) v = resid[o] + gamma[col] * v;
        if (outH) outH[o] = (_Float16)v;
        else      outF[o] = v;
      }
    }
}

// ---------------------------------------------------------------------------
// Flash-style attention: one block per (b, h, 128 q-rows). 4 waves.
// q,k,v f16 (kv packed [k|v] per row). Online softmax, WMMA for both GEMMs.
// ---------------------------------------------------------------------------
__global__ __launch_bounds__(128) void attn_flash(const _Float16* __restrict__ qh,
                                                  const _Float16* __restrict__ kvh,
                                                  _Float16* __restrict__ oh) {
  __shared__ _Float16 kbuf[32 * 96];      // [kvrow][d]
  __shared__ _Float16 vbuf[96 * 32];      // [d][kvrow] (transposed)
  __shared__ _Float16 pbuf[4][32 * 32];   // per-wave P staging
  const int blk = blockIdx.x;
  const int qt = blk & 7;
  const int h  = (blk >> 3) & 7;
  const int b  = blk >> 6;
  const int tid = threadIdx.x;
  const int w = tid >> 5, l = tid & 31;
  const int l15 = l & 15, lh = (l >> 4) & 1;
  const float scl = 0.1020620726f;        // 1/sqrt(96)

  // Q fragments (A layout): 2 M-tiles x 3 K-chunks of 32
  v16h qf[2][3];
  const size_t qrow0 = (size_t)(b * CN + qt * 128 + w * 32);
#pragma unroll
  for (int mt = 0; mt < 2; ++mt)
#pragma unroll
    for (int kc = 0; kc < 3; ++kc) {
      const _Float16* p = qh + (qrow0 + mt * 16 + l15) * CD + h * CHD + kc * 32 + lh * 16;
      v8h lo = *(const v8h*)p;
      v8h hi = *(const v8h*)(p + 8);
      v16h a;
#pragma unroll
      for (int i = 0; i < 8; ++i) { a[i] = lo[i]; a[i + 8] = hi[i]; }
      qf[mt][kc] = a;
    }

  float mrow[2][8], lrow[2][8];
  v8f O[2][6];
#pragma unroll
  for (int mt = 0; mt < 2; ++mt)
#pragma unroll
    for (int r = 0; r < 8; ++r) { mrow[mt][r] = -3.0e38f; lrow[mt][r] = 0.f; }
#pragma unroll
  for (int mt = 0; mt < 2; ++mt)
#pragma unroll
    for (int nt = 0; nt < 6; ++nt)
#pragma unroll
      for (int r = 0; r < 8; ++r) O[mt][nt][r] = 0.f;

  for (int kb = 0; kb < 32; ++kb) {
    const size_t kvrow0 = (size_t)(b * CN + kb * 32);
    // K tile (32x96 halves) via async DMA: 384 b128 chunks over 128 threads
    for (int c = tid; c < 384; c += 128) {
      int r = c / 12, s = c % 12;
      async_copy_b128(kvh + (kvrow0 + r) * (2 * CD) + h * CHD + s * 8,
                      kbuf + r * 96 + s * 8);
    }
    // V tile, vector loads + transposed LDS scatter
    for (int c = tid; c < 384; c += 128) {
      int r = c / 12, s = c % 12;
      v8h vv = *(const v8h*)(kvh + (kvrow0 + r) * (2 * CD) + CD + h * CHD + s * 8);
#pragma unroll
      for (int i = 0; i < 8; ++i) vbuf[(s * 8 + i) * 32 + r] = vv[i];
    }
    wait_async0();
    __syncthreads();

    // S = q * K^T  (2x2 16x16 tiles per wave, K=96 in 3 steps)
    v8f S[2][2];
#pragma unroll
    for (int mt = 0; mt < 2; ++mt)
#pragma unroll
      for (int nt = 0; nt < 2; ++nt)
#pragma unroll
        for (int r = 0; r < 8; ++r) S[mt][nt][r] = 0.f;
#pragma unroll
    for (int kc = 0; kc < 3; ++kc) {
      v16h bf[2];
#pragma unroll
      for (int nt = 0; nt < 2; ++nt)
        bf[nt] = *(const v16h*)(kbuf + (nt * 16 + l15) * 96 + kc * 32 + lh * 16);
#pragma unroll
      for (int mt = 0; mt < 2; ++mt)
#pragma unroll
        for (int nt = 0; nt < 2; ++nt)
          S[mt][nt] = __builtin_amdgcn_wmma_f32_16x16x32_f16(
              false, qf[mt][kc], false, bf[nt], (short)0, S[mt][nt], false, false);
    }

    // online softmax update (rows live across 16-lane half-groups)
    float sc[2][8];
#pragma unroll
    for (int mt = 0; mt < 2; ++mt)
#pragma unroll
      for (int r = 0; r < 8; ++r) {
        float s0 = S[mt][0][r] * scl;
        float s1 = S[mt][1][r] * scl;
        float v = fmaxf(s0, s1);
        v = fmaxf(v, __shfl_xor(v, 1, 32));
        v = fmaxf(v, __shfl_xor(v, 2, 32));
        v = fmaxf(v, __shfl_xor(v, 4, 32));
        v = fmaxf(v, __shfl_xor(v, 8, 32));
        float mn = fmaxf(mrow[mt][r], v);
        float alpha = __expf(mrow[mt][r] - mn);
        mrow[mt][r] = mn;
        float p0 = __expf(s0 - mn);
        float p1 = __expf(s1 - mn);
        S[mt][0][r] = p0; S[mt][1][r] = p1;
        float rs = p0 + p1;
        rs += __shfl_xor(rs, 1, 32);
        rs += __shfl_xor(rs, 2, 32);
        rs += __shfl_xor(rs, 4, 32);
        rs += __shfl_xor(rs, 8, 32);
        lrow[mt][r] = lrow[mt][r] * alpha + rs;
        sc[mt][r] = alpha;
      }

    // stage P (C layout -> LDS), per-wave region so no extra barrier needed
#pragma unroll
    for (int mt = 0; mt < 2; ++mt)
#pragma unroll
      for (int nt = 0; nt < 2; ++nt)
#pragma unroll
        for (int r = 0; r < 8; ++r)
          pbuf[w][(mt * 16 + r + lh * 8) * 32 + nt * 16 + l15] = (_Float16)S[mt][nt][r];

    // rescale O accumulators
#pragma unroll
    for (int mt = 0; mt < 2; ++mt)
#pragma unroll
      for (int nt = 0; nt < 6; ++nt)
#pragma unroll
        for (int r = 0; r < 8; ++r) O[mt][nt][r] *= sc[mt][r];

    // O += P * V (K=32 single WMMA step, 6 d-tiles)
    v16h pf[2];
#pragma unroll
    for (int mt = 0; mt < 2; ++mt) {
      const _Float16* p = &pbuf[w][(mt * 16 + l15) * 32 + lh * 16];
      v8h lo = *(const v8h*)p;
      v8h hi = *(const v8h*)(p + 8);
      v16h a;
#pragma unroll
      for (int i = 0; i < 8; ++i) { a[i] = lo[i]; a[i + 8] = hi[i]; }
      pf[mt] = a;
    }
#pragma unroll
    for (int nt = 0; nt < 6; ++nt) {
      v16h vf = *(const v16h*)(vbuf + (nt * 16 + l15) * 32 + lh * 16);
#pragma unroll
      for (int mt = 0; mt < 2; ++mt)
        O[mt][nt] = __builtin_amdgcn_wmma_f32_16x16x32_f16(
            false, pf[mt], false, vf, (short)0, O[mt][nt], false, false);
    }
    __syncthreads();
  }

#pragma unroll
  for (int mt = 0; mt < 2; ++mt)
#pragma unroll
    for (int nt = 0; nt < 6; ++nt)
#pragma unroll
      for (int r = 0; r < 8; ++r) {
        size_t row = qrow0 + mt * 16 + r + lh * 8;
        int col = h * CHD + nt * 16 + l15;
        oh[row * CD + col] = (_Float16)(O[mt][nt][r] / lrow[mt][r]);
      }
}

// ---------------------------------------------------------------------------
// Column softmax stats over N per (b, channel): max & sum(exp)
// ---------------------------------------------------------------------------
__global__ __launch_bounds__(256) void colstats(const float* __restrict__ xf,
                                                const _Float16* __restrict__ xh,
                                                float* __restrict__ Mo,
                                                float* __restrict__ So) {
  __shared__ float red[256];
  const int blk = blockIdx.x;
  const int b = blk / CD, c = blk % CD;
  const int tid = threadIdx.x;
  float mx = -3.0e38f;
  for (int n = tid; n < CN; n += 256) {
    size_t idx = (size_t)(b * CN + n) * CD + c;
    float v = xf ? xf[idx] : (float)xh[idx];
    mx = fmaxf(mx, v);
  }
  red[tid] = mx; __syncthreads();
  for (int s = 128; s > 0; s >>= 1) { if (tid < s) red[tid] = fmaxf(red[tid], red[tid + s]); __syncthreads(); }
  mx = red[0]; __syncthreads();
  float sm = 0.f;
  for (int n = tid; n < CN; n += 256) {
    size_t idx = (size_t)(b * CN + n) * CD + c;
    float v = xf ? xf[idx] : (float)xh[idx];
    sm += __expf(v - mx);
  }
  red[tid] = sm; __syncthreads();
  for (int s = 128; s > 0; s >>= 1) { if (tid < s) red[tid] += red[tid + s]; __syncthreads(); }
  if (tid == 0) { Mo[blk] = mx; So[blk] = red[0]; }
}

// khat[b,h,n] = mean_d softmax_N(x2n)[n, h*96+d]; one wave per (b,h,n)
__global__ __launch_bounds__(32) void khat_kernel(const _Float16* __restrict__ x2n,
                                                  const float* __restrict__ Mx,
                                                  const float* __restrict__ Sx,
                                                  float* __restrict__ khat) {
  const int blk = blockIdx.x;
  const int n = blk & (CN - 1);
  const int h = (blk >> 10) & 7;
  const int b = blk >> 13;
  const int l = threadIdx.x;
  float s = 0.f;
  for (int d = l; d < CHD; d += 32) {
    int ch = h * CHD + d;
    s += __expf((float)x2n[(size_t)(b * CN + n) * CD + ch] - Mx[b * CD + ch]) / Sx[b * CD + ch];
  }
  s += __shfl_xor(s, 1, 32);
  s += __shfl_xor(s, 2, 32);
  s += __shfl_xor(s, 4, 32);
  s += __shfl_xor(s, 8, 32);
  s += __shfl_xor(s, 16, 32);
  if (l == 0) khat[(size_t)(b * CH + h) * CN + n] = s * (1.f / CHD);
}

// attn[b,c] = sigmoid( sum_n softmax_N(cpq)[n,c] * khat[b,h(c),n] ) * temp[h]
__global__ __launch_bounds__(256) void attnvec_kernel(const float* __restrict__ cpq,
                                                      const float* __restrict__ Mq,
                                                      const float* __restrict__ Sq,
                                                      const float* __restrict__ khat,
                                                      const float* __restrict__ temp,
                                                      float* __restrict__ av) {
  __shared__ float red[256];
  const int blk = blockIdx.x;
  const int b = blk / CD, c = blk % CD, h = c / CHD;
  const int tid = threadIdx.x;
  const float M = Mq[blk], Sv = Sq[blk];
  float s = 0.f;
  for (int n = tid; n < CN; n += 256)
    s += __expf(cpq[(size_t)(b * CN + n) * CD + c] - M) / Sv *
         khat[(size_t)(b * CH + h) * CN + n];
  red[tid] = s; __syncthreads();
  for (int st = 128; st > 0; st >>= 1) { if (tid < st) red[tid] += red[tid + st]; __syncthreads(); }
  if (tid == 0) av[blk] = (1.f / (1.f + __expf(-red[0]))) * temp[h];
}

// depthwise 3x3 (SAME, zero pad) + bias + exact GELU
__global__ __launch_bounds__(256) void dwconv_gelu(const _Float16* __restrict__ h1,
                                                   const float* __restrict__ dww,
                                                   const float* __restrict__ dwb,
                                                   _Float16* __restrict__ h2) {
  size_t idx = (size_t)blockIdx.x * 256 + threadIdx.x;
  if (idx >= (size_t)CB * CN * CHID) return;
  int c = (int)(idx % CHID);
  int n = (int)((idx / CHID) % CN);
  int b = (int)(idx / ((size_t)CHID * CN));
  int y = n >> 5, x0 = n & 31;
  float acc = dwb[c];
#pragma unroll
  for (int dy = 0; dy < 3; ++dy) {
    int yy = y + dy - 1;
    if (yy < 0 || yy > 31) continue;
#pragma unroll
    for (int dx = 0; dx < 3; ++dx) {
      int xx = x0 + dx - 1;
      if (xx < 0 || xx > 31) continue;
      acc += (float)h1[((size_t)(b * CN) + (yy << 5) + xx) * CHID + c] * dww[c * 9 + dy * 3 + dx];
    }
  }
  float g = 0.5f * acc * (1.f + erff(acc * 0.70710678f));
  h2[idx] = (_Float16)g;
}

// LN(vraw) then out = x2 + gamma2 * attn[b,c] * vmlp
__global__ __launch_bounds__(256) void final_out(const float* __restrict__ vraw,
                                                 const float* __restrict__ nvw,
                                                 const float* __restrict__ nvb,
                                                 const float* __restrict__ av,
                                                 const float* __restrict__ g2,
                                                 const float* __restrict__ x2,
                                                 float* __restrict__ out) {
  __shared__ float red[256];
  const int row = blockIdx.x, tid = threadIdx.x;
  const int b = row >> 10;
  const float* vr = vraw + (size_t)row * CD;
  float s = 0.f, s2 = 0.f;
  for (int c = tid; c < CD; c += 256) { float v = vr[c]; s += v; s2 += v * v; }
  red[tid] = s; __syncthreads();
  for (int st = 128; st > 0; st >>= 1) { if (tid < st) red[tid] += red[tid + st]; __syncthreads(); }
  float mean = red[0] / CD; __syncthreads();
  red[tid] = s2; __syncthreads();
  for (int st = 128; st > 0; st >>= 1) { if (tid < st) red[tid] += red[tid + st]; __syncthreads(); }
  float var = red[0] / CD - mean * mean;
  float rs = rsqrtf(var + 1e-5f);
  for (int c = tid; c < CD; c += 256) {
    float vm = (vr[c] - mean) * rs * nvw[c] + nvb[c];
    out[(size_t)row * CD + c] = x2[(size_t)row * CD + c] + g2[c] * av[b * CD + c] * vm;
  }
}

// ---------------------------------------------------------------------------
extern "C" void kernel_launch(void* const* d_in, const int* in_sizes, int n_in,
                              void* d_out, int out_size, void* d_ws, size_t ws_size,
                              hipStream_t stream) {
  (void)in_sizes; (void)n_in; (void)out_size; (void)ws_size;
  const float* x      = (const float*)d_in[0];
  const float* n1w    = (const float*)d_in[1];
  const float* n1b    = (const float*)d_in[2];
  const float* q_w    = (const float*)d_in[3];
  const float* kv_w   = (const float*)d_in[4];
  const float* proj_w = (const float*)d_in[5];
  const float* proj_b = (const float*)d_in[6];
  const float* gamma1 = (const float*)d_in[7];
  const float* n2w    = (const float*)d_in[8];
  const float* n2b    = (const float*)d_in[9];
  const float* cpq_w  = (const float*)d_in[10];
  const float* temp   = (const float*)d_in[11];
  const float* fc1_w  = (const float*)d_in[12];
  const float* fc1_b  = (const float*)d_in[13];
  const float* dw_w   = (const float*)d_in[14];
  const float* dw_b   = (const float*)d_in[15];
  const float* fc2_w  = (const float*)d_in[16];
  const float* fc2_b  = (const float*)d_in[17];
  const float* nvw    = (const float*)d_in[18];
  const float* nvb    = (const float*)d_in[19];
  const float* gamma2 = (const float*)d_in[20];
  float* out = (float*)d_out;

  char* base = (char*)d_ws;
  size_t off = 0;
  auto alloc = [&](size_t bytes) -> char* {
    char* p = base + off;
    off += (bytes + 255) & ~(size_t)255;
    return p;
  };

  _Float16* wq   = (_Float16*)alloc((size_t)CD * CD * 2);
  _Float16* wkv  = (_Float16*)alloc((size_t)2 * CD * CD * 2);
  _Float16* wpj  = (_Float16*)alloc((size_t)CD * CD * 2);
  _Float16* wcq  = (_Float16*)alloc((size_t)CD * CD * 2);
  _Float16* wf1  = (_Float16*)alloc((size_t)CHID * CD * 2);
  _Float16* wf2  = (_Float16*)alloc((size_t)CD * CHID * 2);
  _Float16* xnH  = (_Float16*)alloc((size_t)CROWS * CD * 2);       // reused as attn_out
  _Float16* qH   = (_Float16*)alloc((size_t)CROWS * CD * 2);       // reused as x2n
  _Float16* kvH  = (_Float16*)alloc((size_t)CROWS * 2 * CD * 2);   // reused as cpq (f32)
  float* x2F     = (float*)alloc((size_t)CROWS * CD * 4);
  float* Mq      = (float*)alloc((size_t)CB * CD * 4);
  float* Sq      = (float*)alloc((size_t)CB * CD * 4);
  float* Mx      = (float*)alloc((size_t)CB * CD * 4);
  float* Sx      = (float*)alloc((size_t)CB * CD * 4);
  float* khat    = (float*)alloc((size_t)CB * CH * CN * 4);
  float* av      = (float*)alloc((size_t)CB * CD * 4);
  _Float16* h1H  = (_Float16*)alloc((size_t)CROWS * CHID * 2);
  _Float16* h2H  = (_Float16*)alloc((size_t)CROWS * CHID * 2);
  float* vrawF   = (float*)alloc((size_t)CROWS * CD * 4);

  _Float16* attnH = xnH;   // reuse: xn dead after q/kv GEMMs
  _Float16* x2nH  = qH;    // reuse: q dead after attention
  float*    cpqF  = (float*)kvH;  // reuse: kv dead after attention (same byte size)

  auto cv = [&](const float* s, _Float16* d, size_t n) {
    cvt_f32_f16<<<(unsigned)((n + 255) / 256), 256, 0, stream>>>(s, d, (int)n);
  };
  cv(q_w,    wq,  (size_t)CD * CD);
  cv(kv_w,   wkv, (size_t)2 * CD * CD);
  cv(proj_w, wpj, (size_t)CD * CD);
  cv(cpq_w,  wcq, (size_t)CD * CD);
  cv(fc1_w,  wf1, (size_t)CHID * CD);
  cv(fc2_w,  wf2, (size_t)CD * CHID);

  // token mixing
  ln_rows<<<CROWS, 256, 0, stream>>>(x, n1w, n1b, xnH);
  gemm_wmma<<<(CROWS / 128) * (CD / 128), 256, 0, stream>>>(
      xnH, wq, CROWS, CD, CD, nullptr, nullptr, nullptr, qH, nullptr);
  gemm_wmma<<<(CROWS / 128) * (2 * CD / 128), 256, 0, stream>>>(
      xnH, wkv, CROWS, 2 * CD, CD, nullptr, nullptr, nullptr, kvH, nullptr);
  attn_flash<<<CB * CH * (CN / 128), 128, 0, stream>>>(qH, kvH, attnH);
  gemm_wmma<<<(CROWS / 128) * (CD / 128), 256, 0, stream>>>(
      attnH, wpj, CROWS, CD, CD, proj_b, x, gamma1, nullptr, x2F);

  // channel processing
  ln_rows<<<CROWS, 256, 0, stream>>>(x2F, n2w, n2b, x2nH);
  gemm_wmma<<<(CROWS / 128) * (CD / 128), 256, 0, stream>>>(
      x2nH, wcq, CROWS, CD, CD, nullptr, nullptr, nullptr, nullptr, cpqF);
  colstats<<<CB * CD, 256, 0, stream>>>(cpqF, nullptr, Mq, Sq);
  colstats<<<CB * CD, 256, 0, stream>>>(nullptr, x2nH, Mx, Sx);
  khat_kernel<<<CB * CH * CN, 32, 0, stream>>>(x2nH, Mx, Sx, khat);
  attnvec_kernel<<<CB * CD, 256, 0, stream>>>(cpqF, Mq, Sq, khat, temp, av);

  // mlp_v branch
  gemm_wmma<<<(CROWS / 128) * (CHID / 128), 256, 0, stream>>>(
      x2nH, wf1, CROWS, CHID, CD, fc1_b, nullptr, nullptr, h1H, nullptr);
  dwconv_gelu<<<(unsigned)(((size_t)CROWS * CHID + 255) / 256), 256, 0, stream>>>(
      h1H, dw_w, dw_b, h2H);
  gemm_wmma<<<(CROWS / 128) * (CD / 128), 256, 0, stream>>>(
      h2H, wf2, CROWS, CD, CHID, fc2_b, nullptr, nullptr, nullptr, vrawF);

  // LN + channel gate + residual
  final_out<<<CROWS, 256, 0, stream>>>(vrawF, nvw, nvb, av, gamma2, x2F, out);
}
